// GNNClassifier_85856396247086
// MI455X (gfx1250) — matrix operand order repaired and checked by reading, not compiled, not run on hardware
//
#include <hip/hip_runtime.h>

typedef __attribute__((ext_vector_type(16))) _Float16 v16h;
typedef __attribute__((ext_vector_type(8)))  _Float16 v8h;
typedef __attribute__((ext_vector_type(8)))  float    v8f;

#define HOUT 64   // hidden width of both GCN layers
#define LDK  136  // LDS K-stride in halves: 128 + 8 pad -> conflict-free b128 reads

// ---------------- elementwise / setup kernels ----------------

__global__ void zero_f32(float* __restrict__ p, int n) {
    int i = blockIdx.x * blockDim.x + threadIdx.x;
    if (i < n) p[i] = 0.0f;
}

__global__ void deg_count(const int* __restrict__ dst, float* __restrict__ deg, int E) {
    int e = blockIdx.x * blockDim.x + threadIdx.x;
    if (e < E) atomicAdd(&deg[dst[e]], 1.0f);
}

__global__ void deg_to_dinv(float* __restrict__ d, int n) {
    int i = blockIdx.x * blockDim.x + threadIdx.x;
    if (i < n) d[i] = rsqrtf(d[i] + 1.0f);   // +1 for self-loop
}

// agg[i][j] = dinv[i]^2 * hlin[i][j] + bias[j]   (self-loop term + bias)
__global__ void init_agg(const float* __restrict__ hlin, const float* __restrict__ dinv,
                         const float* __restrict__ bias, float* __restrict__ agg, int Nn) {
    int gid = blockIdx.x * blockDim.x + threadIdx.x;
    int i = gid >> 6, j = gid & 63;
    if (i >= Nn) return;
    float di = dinv[i];
    agg[gid] = di * di * hlin[gid] + bias[j];
}

// one thread per (edge, channel); 64 consecutive lanes share an edge so the
// gather of hlin[src] and the atomics to agg[dst] are cacheline-coherent
__global__ void edge_scatter(const float* __restrict__ hlin, const int* __restrict__ src,
                             const int* __restrict__ dst, const float* __restrict__ dinv,
                             float* __restrict__ agg, int E) {
    int gid = blockIdx.x * blockDim.x + threadIdx.x;
    int e = gid >> 6, j = gid & 63;
    if (e >= E) return;
    int s = src[e], d = dst[e];
    float norm = dinv[s] * dinv[d];
    atomicAdd(&agg[d * HOUT + j], norm * hlin[s * HOUT + j]);
}

// ---------------- WMMA GEMM: C[nrows x 64] = act(A[nrows x K]) * B[K x 64] ------------
// f32 in memory -> f16 staged in LDS once per block, f32 accumulate.
// 256 threads = 8 waves; each wave owns a 16x16 tile; block covers 32 rows x 64 cols.
template <int K, bool RELU>
__global__ __launch_bounds__(256) void gemm_wmma64(const float* __restrict__ A,
                                                   const float* __restrict__ B,
                                                   float* __restrict__ C, int nrows) {
    __shared__ _Float16 lA[32][LDK];   // row-major   [m][k]
    __shared__ _Float16 lB[HOUT][LDK]; // col-major   [n][k]

    const int m0blk = blockIdx.x * 32;

    // stage A (coalesced along k), ReLU fused at convert time
    for (int idx = threadIdx.x; idx < 32 * K; idx += 256) {
        int m = idx / K, k = idx & (K - 1);
        int row = min(m0blk + m, nrows - 1);
        float v = A[(size_t)row * K + k];
        if (RELU) v = fmaxf(v, 0.0f);
        lA[m][k] = (_Float16)v;
    }
    // stage B (coalesced along n), transpose to col-major
    for (int idx = threadIdx.x; idx < HOUT * K; idx += 256) {
        int k = idx >> 6, n = idx & 63;
        lB[n][k] = (_Float16)B[(size_t)k * HOUT + n];
    }
    __syncthreads();

    const int lane = threadIdx.x & 31;
    const int wave = threadIdx.x >> 5;
    const int lm = lane & 15;
    const int lh = lane >> 4;
    const int mloc = (wave >> 2) * 16;  // wave's row tile within block
    const int n0   = (wave & 3) * 16;   // wave's col tile

    const _Float16* pa = &lA[mloc + lm][0];
    const _Float16* pb = &lB[n0 + lm][0];

    v8f acc = {};
#pragma unroll
    for (int kt = 0; kt < K; kt += 32) {
        // A fragment: lane holds K = kt + lh*8 + {0..7} and + {16..23} (two b128 runs)
        v8h a0 = *reinterpret_cast<const v8h*>(pa + kt + lh * 8);
        v8h a1 = *reinterpret_cast<const v8h*>(pa + kt + lh * 8 + 16);
        v16h a = __builtin_shufflevector(a0, a1, 0, 1, 2, 3, 4, 5, 6, 7,
                                                 8, 9, 10, 11, 12, 13, 14, 15);
        // B fragment: lane holds K = kt + lh*8 + {0..7} and + {16..23} (two b128 runs)
        v8h b0 = *reinterpret_cast<const v8h*>(pb + kt + lh * 8);
        v8h b1 = *reinterpret_cast<const v8h*>(pb + kt + lh * 8 + 16);
        v16h b = __builtin_shufflevector(b0, b1, 0, 1, 2, 3, 4, 5, 6, 7,
                                                 8, 9, 10, 11, 12, 13, 14, 15);
        acc = __builtin_amdgcn_wmma_f32_16x16x32_f16(
            /*neg_a=*/false, a, /*neg_b=*/false, b,
            /*c_mod=*/(short)0, acc, /*reuse_a=*/false, /*reuse_b=*/false);
    }

    // C/D layout: row = r + 8*lh, col = lm
    const int m0w = m0blk + mloc;
    float* cp = C + (size_t)(m0w + 8 * lh) * HOUT + n0 + lm;
    if (m0w + 16 <= nrows) {           // wave-uniform fast path: clean stores
#pragma unroll
        for (int r = 0; r < 8; ++r) cp[(size_t)r * HOUT] = acc[r];
    } else {                            // tail (unused for N % 32 == 0)
#pragma unroll
        for (int r = 0; r < 8; ++r)
            if (m0w + 8 * lh + r < nrows) cp[(size_t)r * HOUT] = acc[r];
    }
}

// ---------------- pooling + classifier ----------------

// layer-2 ReLU fused here
__global__ void pool_add(const float* __restrict__ h, const int* __restrict__ batch,
                         float* __restrict__ sums, float* __restrict__ cnt, int Nn) {
    int gid = blockIdx.x * blockDim.x + threadIdx.x;
    int i = gid >> 6, j = gid & 63;
    if (i >= Nn) return;
    int g = batch[i];
    atomicAdd(&sums[g * HOUT + j], fmaxf(h[gid], 0.0f));
    if (j == 0) atomicAdd(&cnt[g], 1.0f);
}

__global__ void classify(const float* __restrict__ sums, const float* __restrict__ cnt,
                         const float* __restrict__ Wc1, const float* __restrict__ bc1,
                         const float* __restrict__ Wc2, const float* __restrict__ bc2,
                         float* __restrict__ out, int G) {
    int g = blockIdx.x * blockDim.x + threadIdx.x;
    if (g >= G) return;
    float inv = 1.0f / fmaxf(cnt[g], 1.0f);
    float o0 = bc2[0], o1 = bc2[1];
    for (int k = 0; k < 32; ++k) {
        float s = bc1[k];
        for (int j = 0; j < 64; ++j)
            s += sums[g * HOUT + j] * inv * Wc1[j * 32 + k];
        s = fmaxf(s, 0.0f);
        o0 += s * Wc2[k * 2 + 0];
        o1 += s * Wc2[k * 2 + 1];
    }
    out[g * 2 + 0] = o0;
    out[g * 2 + 1] = o1;
}

// ---------------- launch ----------------

extern "C" void kernel_launch(void* const* d_in, const int* in_sizes, int n_in,
                              void* d_out, int out_size, void* d_ws, size_t ws_size,
                              hipStream_t stream) {
    const float* x     = (const float*)d_in[0];
    const int*   ei    = (const int*)  d_in[1];
    const int*   batch = (const int*)  d_in[2];
    const float* W1    = (const float*)d_in[3];
    const float* b1    = (const float*)d_in[4];
    const float* W2    = (const float*)d_in[5];
    const float* b2    = (const float*)d_in[6];
    const float* Wc1   = (const float*)d_in[7];
    const float* bc1   = (const float*)d_in[8];
    const float* Wc2   = (const float*)d_in[9];
    const float* bc2   = (const float*)d_in[10];
    float* out = (float*)d_out;

    const int Nn = in_sizes[2];          // 100000 nodes
    const int E  = in_sizes[1] / 2;      // 1.6M edges
    const int G  = out_size / 2;         // 256 graphs

    const int* src = ei;
    const int* dst = ei + E;

    // workspace: bufA | bufB | dinv | sums | cnt   (~52 MB)
    float* ws   = (float*)d_ws;
    size_t nh   = (size_t)Nn * HOUT;
    float* bufA = ws;
    float* bufB = bufA + nh;
    float* dinv = bufB + nh;
    float* sums = dinv + Nn;
    float* cnt  = sums + (size_t)G * HOUT;

    auto cdiv = [](long long a, long long b) { return (int)((a + b - 1) / b); };
    dim3 blk(256);
    const int nhBlocks = cdiv((long long)Nn * HOUT, 256);
    const int ehBlocks = cdiv((long long)E * HOUT, 256);
    const int gBlocks  = cdiv(Nn, 32);

    // degree -> dinv
    zero_f32   <<<cdiv(Nn, 256), blk, 0, stream>>>(dinv, Nn);
    deg_count  <<<cdiv(E, 256),  blk, 0, stream>>>(dst, dinv, E);
    deg_to_dinv<<<cdiv(Nn, 256), blk, 0, stream>>>(dinv, Nn);

    // layer 1: lin(WMMA) -> self-loop+bias -> scatter   (ReLU deferred into layer-2 GEMM)
    gemm_wmma64<128, false><<<gBlocks,  blk, 0, stream>>>(x, W1, bufA, Nn);
    init_agg              <<<nhBlocks, blk, 0, stream>>>(bufA, dinv, b1, bufB, Nn);
    edge_scatter          <<<ehBlocks, blk, 0, stream>>>(bufA, src, dst, dinv, bufB, E);

    // layer 2: relu fused into A-staging of the GEMM
    gemm_wmma64<64, true><<<gBlocks,  blk, 0, stream>>>(bufB, W2, bufA, Nn);
    init_agg             <<<nhBlocks, blk, 0, stream>>>(bufA, dinv, b2, bufB, Nn);
    edge_scatter         <<<ehBlocks, blk, 0, stream>>>(bufA, src, dst, dinv, bufB, E);

    // global mean pool (ReLU fused) + MLP head
    zero_f32 <<<cdiv((long long)G * HOUT + G, 256), blk, 0, stream>>>(sums, G * HOUT + G);
    pool_add <<<nhBlocks, blk, 0, stream>>>(bufB, batch, sums, cnt, Nn);
    classify <<<cdiv(G, 256), blk, 0, stream>>>(sums, cnt, Wc1, bc1, Wc2, bc2, out, G);
}
